// MonotonicNN_3753801417178
// MI455X (gfx1250) — matrix operand order repaired and checked by reading, not compiled, not run on hardware
//
#include <hip/hip_runtime.h>

typedef __attribute__((ext_vector_type(16))) _Float16 v16h;
typedef __attribute__((ext_vector_type(8)))  float    v8f;

#define NPTS 51
#define HD   256
#define B1S  260              // base1 row stride in floats (260%64=4 -> conflict-free lane spread)
#define ACTS 264              // act row stride in halves
#define SMEM_BYTES (131072 + (16*B1S + 256 + 256 + 256 + 16 + 128) * 4)
#define DPI  3.14159265358979323846

// A-matrix (16x32 f16) per-lane K index for half h (contiguous: K = 16*(lane/16) + h, matches B packing)
__device__ __forceinline__ int amapK(int lane, int hh) {
  return (lane < 16) ? hh : hh + 16;
}

// ---------------------------------------------------------------------------
// Pack kernel: f32 weights -> f16 B-fragments in ws; also cc/steps weights.
// B-fragment layout (32x16 f16): lane<16: N=lane, K=h ; lane>=16: N=lane-16, K=16+h
// storage: frag-major, ((frag*32 + lane)*16 + h)
// ---------------------------------------------------------------------------
__global__ void mnn_pack_kernel(const float* __restrict__ iw1, const float* __restrict__ iw2,
                                const float* __restrict__ nw1, const float* __restrict__ nw2,
                                _Float16* __restrict__ w2p, _Float16* __restrict__ w1p,
                                _Float16* __restrict__ nw2p, _Float16* __restrict__ nw1p,
                                float* __restrict__ ccs, float* __restrict__ steps) {
  int idx = blockIdx.x * 256 + threadIdx.x;
  if (idx < 65536) {                       // iw2: 8 k_tiles x 16 n_tiles
    int e = idx, hh = e & 15, lane = (e >> 4) & 31, lin = e >> 9;
    int nt = lin & 15, kt = lin >> 4;
    int K = kt * 32 + ((lane < 16) ? hh : 16 + hh);
    int N = nt * 16 + (lane & 15);
    w2p[e] = (_Float16)iw2[N * 256 + K];
  } else if (idx < 73728) {                // iw1: K(z)=16 padded to 32
    int e = idx - 65536, hh = e & 15, lane = (e >> 4) & 31, nt = e >> 9;
    int K = (lane < 16) ? hh : 16 + hh;
    int N = nt * 16 + (lane & 15);
    w1p[e] = (_Float16)((K < 16) ? iw1[N * 16 + K] : 0.f);
  } else if (idx < 139264) {               // nw2
    int e = idx - 73728, hh = e & 15, lane = (e >> 4) & 31, lin = e >> 9;
    int nt = lin & 15, kt = lin >> 4;
    int K = kt * 32 + ((lane < 16) ? hh : 16 + hh);
    int N = nt * 16 + (lane & 15);
    nw2p[e] = (_Float16)nw2[N * 256 + K];
  } else if (idx < 147456) {               // nw1: K=15 padded
    int e = idx - 139264, hh = e & 15, lane = (e >> 4) & 31, nt = e >> 9;
    int K = (lane < 16) ? hh : 16 + hh;
    int N = nt * 16 + (lane & 15);
    nw1p[e] = (_Float16)((K < 15) ? nw1[N * 15 + K] : 0.f);
  } else if (idx < 147456 + NPTS) {        // Clenshaw-Curtis weights
    int i = idx - 147456;
    double acc = 0.0;
    for (int j = 0; j <= 50; j += 2) {
      double Wv = (j == 0) ? 1.0 : 2.0 / (1.0 - (double)(j * j));
      double g = (i == 0) ? 0.5 : cos((double)j * (double)i * DPI / 50.0) * ((i == 50) ? 0.5 : 1.0);
      acc += g * Wv;
    }
    ccs[i] = (float)(acc * (2.0 / 50.0));
  } else if (idx < 147456 + 2 * NPTS) {
    int i = idx - 147456 - NPTS;
    steps[i] = (float)cos((double)i * DPI / 50.0);
  }
}

// ---------------------------------------------------------------------------
// Integrand kernel: 512 blocks x 256 threads (8 waves). Block = 16 batch rows.
// base1 computed once (rank-1 x-update per quadrature point), W2 staged in LDS
// via CDNA5 async cache->LDS DMA (ASYNCcnt), overlapped with wave-0's WMMAs.
// ---------------------------------------------------------------------------
__global__ void __launch_bounds__(256) mnn_integrand_kernel(
    const float* __restrict__ x, const float* __restrict__ h,
    const float* __restrict__ iw1, const float* __restrict__ ib1,
    const float* __restrict__ ib2, const float* __restrict__ iw3,
    const float* __restrict__ ib3,
    const _Float16* __restrict__ w2p, const _Float16* __restrict__ w1p,
    const float* __restrict__ ccs, const float* __restrict__ steps,
    float* __restrict__ integ) {
  extern __shared__ char smem[];
  _Float16* w2lds = (_Float16*)smem;                    // 65536 halves (128 KB)
  float* base1 = (float*)(smem + 131072);               // 16 x B1S
  float* w1c0 = base1 + 16 * B1S;                       // 256
  float* b2s = w1c0 + 256;                              // 256
  float* w3s = b2s + 256;                               // 256
  float* xs = w3s + 256;                                // 16
  float* partialW = xs + 16;                            // 8 x 16

  const int tid = threadIdx.x, lane = tid & 31, wave = tid >> 5;
  const int b0 = blockIdx.x * 16;
  const int M = lane & 15;

  // Stage W2 fragments into LDS with async DMA (no VGPR round-trip; ASYNCcnt).
  // Pack order == LDS order, so it is a straight 128 KB block copy.
  {
    unsigned ldsBase = (unsigned)(uintptr_t)w2lds;   // low 32 bits of flat = LDS byte addr
    unsigned goff = (unsigned)tid * 16u;
    unsigned long long gbase = (unsigned long long)(uintptr_t)w2p;
    #pragma unroll
    for (int i = 0; i < 32; ++i) {
      unsigned lds_addr = ldsBase + goff + (unsigned)i * 4096u;
      unsigned gvoff = goff + (unsigned)i * 4096u;
      asm volatile("global_load_async_to_lds_b128 %0, %1, %2 offset:0"
                   :: "v"(lds_addr), "v"(gvoff), "s"(gbase)
                   : "memory");
    }
  }
  w1c0[tid] = iw1[tid * 16];    // column 0 of iw1
  b2s[tid] = ib2[tid];
  w3s[tid] = iw3[tid];
  if (tid < 16) xs[tid] = x[b0 + tid];

  // Wave 0: base1 = [0,h] @ W1^T + b1   (16 x 256, f32 pre-activation)
  // Runs concurrently with the async W2 DMA above.
  if (wave == 0) {
    v16h a;
    #pragma unroll
    for (int hh = 0; hh < 16; ++hh) {
      int K = amapK(lane, hh);
      float v = (K >= 1 && K < 16) ? h[(b0 + M) * 15 + (K - 1)] : 0.f;
      a[hh] = (_Float16)v;
    }
    const int mb = (lane < 16) ? 0 : 8;
    for (int nt = 0; nt < 16; ++nt) {
      v8f c = {};
      v16h bf = *(const v16h*)(w1p + (nt * 32 + lane) * 16);
      c = __builtin_amdgcn_wmma_f32_16x16x32_f16(false, a, false, bf, (short)0, c, false, false);
      int n = nt * 16 + M;
      float b1n = ib1[n];
      #pragma unroll
      for (int r = 0; r < 8; ++r) base1[(r + mb) * B1S + n] = c[r] + b1n;
    }
  }
  asm volatile("s_wait_asynccnt 0" ::: "memory");   // W2 DMA complete
  __syncthreads();

  const float ib3v = ib3[0];
  const float xmb = xs[M];
  float fsum[8];
  #pragma unroll
  for (int r = 0; r < 8; ++r) fsum[r] = 0.f;

  for (int p = wave; p < NPTS; p += 8) {
    const float sp = steps[p];
    const float xm = xmb * (sp + 1.f) * 0.5f;
    // Build relu'd layer-1 activations as A fragments (rank-1 x update)
    v16h af[8];
    #pragma unroll
    for (int kt = 0; kt < 8; ++kt) {
      #pragma unroll
      for (int hh = 0; hh < 16; ++hh) {
        int K = kt * 32 + amapK(lane, hh);
        float v = base1[M * B1S + K] + xm * w1c0[K];
        af[kt][hh] = (_Float16)(v > 0.f ? v : 0.f);
      }
    }
    // Layer 2 (dominant GEMM) fused with layer-3 dot
    float acc[8];
    #pragma unroll
    for (int r = 0; r < 8; ++r) acc[r] = 0.f;
    for (int nt = 0; nt < 16; ++nt) {
      v8f c = {};
      #pragma unroll
      for (int kt = 0; kt < 8; ++kt) {
        v16h bf = *(const v16h*)(w2lds + ((kt * 16 + nt) * 32 + lane) * 16);
        c = __builtin_amdgcn_wmma_f32_16x16x32_f16(false, af[kt], false, bf, (short)0, c, false, false);
      }
      int n = nt * 16 + M;
      float b2n = b2s[n], w3n = w3s[n];
      #pragma unroll
      for (int r = 0; r < 8; ++r) {
        float t = c[r] + b2n;
        t = t > 0.f ? t : 0.f;
        acc[r] += t * w3n;
      }
    }
    const float ccp = ccs[p];
    #pragma unroll
    for (int r = 0; r < 8; ++r) {
      float t = acc[r];
      #pragma unroll
      for (int off = 8; off; off >>= 1) t += __shfl_xor(t, off, 16);
      float u = t + ib3v;
      float f = u > 0.f ? (u + 1.f) : __expf(u);   // elu(u)+1
      fsum[r] += f * ccp;
    }
  }
  if ((lane & 15) == 0) {
    int g = lane >> 4;
    #pragma unroll
    for (int r = 0; r < 8; ++r) partialW[wave * 16 + g * 8 + r] = fsum[r];
  }
  __syncthreads();
  if (tid < 16) {
    float s = 0.f;
    #pragma unroll
    for (int w = 0; w < 8; ++w) s += partialW[w * 16 + tid];
    integ[b0 + tid] = s * xs[tid] * 0.5f;
  }
}

// ---------------------------------------------------------------------------
// Head kernel: offset/scaling MLP (15->256->256->2), 512 single-wave blocks.
// ---------------------------------------------------------------------------
__global__ void __launch_bounds__(32) mnn_head_kernel(
    const float* __restrict__ h, const float* __restrict__ nb1,
    const float* __restrict__ nb2, const float* __restrict__ nw3,
    const float* __restrict__ nb3,
    const _Float16* __restrict__ nw2p, const _Float16* __restrict__ nw1p,
    const float* __restrict__ integ, float* __restrict__ out) {
  __shared__ _Float16 act[16 * ACTS];
  const int lane = threadIdx.x;
  const int b0 = blockIdx.x * 16;
  const int M = lane & 15;
  const int mb = (lane < 16) ? 0 : 8;

  // Layer 1: [h,0] @ nw1^T, ReLU, bounce through LDS (D-layout -> A-layout)
  v16h a;
  #pragma unroll
  for (int hh = 0; hh < 16; ++hh) {
    int K = amapK(lane, hh);
    float v = (K < 15) ? h[(b0 + M) * 15 + K] : 0.f;
    a[hh] = (_Float16)v;
  }
  for (int nt = 0; nt < 16; ++nt) {
    v8f c = {};
    v16h bf = *(const v16h*)(nw1p + (nt * 32 + lane) * 16);
    c = __builtin_amdgcn_wmma_f32_16x16x32_f16(false, a, false, bf, (short)0, c, false, false);
    int n = nt * 16 + M;
    float b1n = nb1[n];
    #pragma unroll
    for (int r = 0; r < 8; ++r) {
      float t = c[r] + b1n;
      act[(r + mb) * ACTS + n] = (_Float16)(t > 0.f ? t : 0.f);
    }
  }
  __syncthreads();

  // Layer 2 fused with both output dots (offset row 0, scaling row 1 of nw3)
  v16h af[8];
  #pragma unroll
  for (int kt = 0; kt < 8; ++kt) {
    #pragma unroll
    for (int hh = 0; hh < 16; ++hh) {
      int K = kt * 32 + amapK(lane, hh);
      af[kt][hh] = act[M * ACTS + K];
    }
  }
  float acco[8], accs[8];
  #pragma unroll
  for (int r = 0; r < 8; ++r) { acco[r] = 0.f; accs[r] = 0.f; }
  for (int nt = 0; nt < 16; ++nt) {
    v8f c = {};
    #pragma unroll
    for (int kt = 0; kt < 8; ++kt) {
      v16h bf = *(const v16h*)(nw2p + ((kt * 16 + nt) * 32 + lane) * 16);
      c = __builtin_amdgcn_wmma_f32_16x16x32_f16(false, af[kt], false, bf, (short)0, c, false, false);
    }
    int n = nt * 16 + M;
    float b2n = nb2[n], wo = nw3[n], wsc = nw3[256 + n];
    #pragma unroll
    for (int r = 0; r < 8; ++r) {
      float t = c[r] + b2n;
      t = t > 0.f ? t : 0.f;
      acco[r] += t * wo;
      accs[r] += t * wsc;
    }
  }
  const float nb3o = nb3[0], nb3s = nb3[1];
  #pragma unroll
  for (int r = 0; r < 8; ++r) {
    float to = acco[r], ts = accs[r];
    #pragma unroll
    for (int off = 8; off; off >>= 1) {
      to += __shfl_xor(to, off, 16);
      ts += __shfl_xor(ts, off, 16);
    }
    if ((lane & 15) == 0) {
      int m = (lane >> 4) * 8 + r;
      float offset = to + nb3o;
      float scaling = __expf(ts + nb3s);
      out[b0 + m] = scaling * integ[b0 + m] + offset;
    }
  }
}

// ---------------------------------------------------------------------------
extern "C" void kernel_launch(void* const* d_in, const int* in_sizes, int n_in,
                              void* d_out, int out_size, void* d_ws, size_t ws_size,
                              hipStream_t stream) {
  const float* x   = (const float*)d_in[0];
  const float* h   = (const float*)d_in[1];
  const float* iw1 = (const float*)d_in[2];
  const float* ib1 = (const float*)d_in[3];
  const float* iw2 = (const float*)d_in[4];
  const float* ib2 = (const float*)d_in[5];
  const float* iw3 = (const float*)d_in[6];
  const float* ib3 = (const float*)d_in[7];
  const float* nw1 = (const float*)d_in[8];
  const float* nb1 = (const float*)d_in[9];
  const float* nw2 = (const float*)d_in[10];
  const float* nb2 = (const float*)d_in[11];
  const float* nw3 = (const float*)d_in[12];
  const float* nb3 = (const float*)d_in[13];
  float* out = (float*)d_out;

  char* ws = (char*)d_ws;
  _Float16* w2p  = (_Float16*)(ws + 0);        // 131072 B
  _Float16* w1p  = (_Float16*)(ws + 131072);   //  16384 B
  _Float16* nw2p = (_Float16*)(ws + 147456);   // 131072 B
  _Float16* nw1p = (_Float16*)(ws + 278528);   //  16384 B
  float* ccs   = (float*)(ws + 294912);        //    204 B
  float* steps = (float*)(ws + 295168);        //    204 B
  float* integ = (float*)(ws + 295424);        //  32768 B

  hipFuncSetAttribute((const void*)mnn_integrand_kernel,
                      hipFuncAttributeMaxDynamicSharedMemorySize, SMEM_BYTES);

  mnn_pack_kernel<<<(147456 + 2 * NPTS + 255) / 256, 256, 0, stream>>>(
      iw1, iw2, nw1, nw2, w2p, w1p, nw2p, nw1p, ccs, steps);

  mnn_integrand_kernel<<<512, 256, SMEM_BYTES, stream>>>(
      x, h, iw1, ib1, ib2, iw3, ib3, w2p, w1p, ccs, steps, integ);

  mnn_head_kernel<<<512, 32, 0, stream>>>(
      h, nb1, nb2, nw3, nb3, nw2p, nw1p, integ, out);
}
